// MambaLayerWrapper_70317204570775
// MI455X (gfx1250) — matrix operand order repaired and checked by reading, not compiled, not run on hardware
//
#include <hip/hip_runtime.h>
#include <hip/hip_bf16.h>
#include <stdint.h>

typedef _Float16 f16;
typedef __attribute__((ext_vector_type(16))) _Float16 v16h;
typedef __attribute__((ext_vector_type(8)))  float    v8f;

#define NB 2
#define HC 96
#define HH 64
#define WV 64
#define LL 4096
#define DI 192
#define NS 16
#define RK 6
#define KD 4
#define BL 8192   // NB*LL

__device__ __forceinline__ float silu_f(float x){ return x / (1.f + expf(-x)); }
__device__ __forceinline__ float sigm_f(float x){ return 1.f / (1.f + expf(-x)); }
__device__ __forceinline__ float gelu_f(float x){ return 0.5f*x*(1.f + erff(x*0.7071067811865476f)); }
__device__ __forceinline__ float softplus_f(float x){ return (x > 20.f) ? x : log1pf(expf(x)); }

// ---------------------------------------------------------------------------
// WMMA helper: C[16x16] += A[16xK] * B[16xK]^T, A/B f16 row-major, one wave.
// A/B use contiguous k order (lane half hi holds k = 16*hi + 0..15), not the ISA table; f32 C/D per ISA.
// ---------------------------------------------------------------------------
union FragH { v16h v; uint32_t u[8]; };

__device__ __forceinline__ v8f wmma_tile(const f16* __restrict__ A, int lda,
                                         const f16* __restrict__ B, int ldb, int K)
{
  const int lane = threadIdx.x & 31;
  const int hi   = lane >> 4;
  const int mn   = lane & 15;
  v8f acc = {};
  for (int kk = 0; kk < K; kk += 32) {
    FragH a, b;
    // A: lanes 0-15 hold M=mn, K = kk + 0..15; lanes 16-31: K = kk + 16..31
    const f16* Ar = A + mn*lda + kk + 16*hi;
#pragma unroll
    for (int p = 0; p < 8; ++p) {
      int ka = 2*p;
      a.u[p] = *(const uint32_t*)(Ar + ka);
    }
    // B: lanes 0-15 hold N=mn, K = kk + 0..15; lanes 16-31: K = kk + 16..31
    const f16* Br = B + mn*ldb + kk + 16*hi;
#pragma unroll
    for (int p = 0; p < 8; ++p)
      b.u[p] = *(const uint32_t*)(Br + 2*p);
    acc = __builtin_amdgcn_wmma_f32_16x16x32_f16(false, a.v, false, b.v,
                                                 (short)0, acc, false, false);
  }
  return acc;
}

// ---------------------------------------------------------------------------
// layout transforms
// ---------------------------------------------------------------------------
__global__ void k_nchw_to_nlc(const float* __restrict__ x, float* __restrict__ xf){
  int i = blockIdx.x*blockDim.x + threadIdx.x;
  if (i >= NB*HC*LL) return;
  int l = i % LL; int c = (i / LL) % HC; int b = i / (HC*LL);
  xf[(b*LL + l)*HC + c] = x[i];
}
__global__ void k_nlc_to_nchw(const float* __restrict__ xf, float* __restrict__ out){
  int i = blockIdx.x*blockDim.x + threadIdx.x;
  if (i >= NB*HC*LL) return;
  int l = i % LL; int c = (i / LL) % HC; int b = i / (HC*LL);
  out[i] = xf[(b*LL + l)*HC + c];
}

// ---------------------------------------------------------------------------
// weight conversion (f32 -> f16, with c-padding for x_proj_w)
// ---------------------------------------------------------------------------
__global__ void k_cvt_f16(const float* __restrict__ src, f16* __restrict__ dst, int n){
  int i = blockIdx.x*blockDim.x + threadIdx.x;
  if (i < n) dst[i] = (f16)src[i];
}
__global__ void k_cvt_xpw(const float* __restrict__ src, f16* __restrict__ dst){
  int i = blockIdx.x*blockDim.x + threadIdx.x;       // over 4*48*192
  if (i >= KD*48*DI) return;
  int d = i % DI; int c = (i/DI) % 48; int k = i / (48*DI);
  dst[i] = (c < RK + 2*NS) ? (f16)src[(k*(RK+2*NS) + c)*DI + d] : (f16)0.f;
}

// ---------------------------------------------------------------------------
// LayerNorm over C=96, one wave per row (3 elements/lane), wave32 shuffles
// ---------------------------------------------------------------------------
__global__ void k_ln96_f16(const float* __restrict__ X, const float* __restrict__ g,
                           const float* __restrict__ bb, f16* __restrict__ out){
  int row  = blockIdx.x*8 + (threadIdx.x >> 5);
  int lane = threadIdx.x & 31;
  const float* xr = X + row*HC;
  float a0 = xr[lane], a1 = xr[lane+32], a2 = xr[lane+64];
  float s = a0+a1+a2, q = a0*a0+a1*a1+a2*a2;
  for (int m = 1; m < 32; m <<= 1){ s += __shfl_xor(s,m,32); q += __shfl_xor(q,m,32); }
  float mu = s*(1.f/HC);
  float rs = rsqrtf(q*(1.f/HC) - mu*mu + 1e-5f);
  out[row*HC + lane]    = (f16)((a0-mu)*rs*g[lane]    + bb[lane]);
  out[row*HC + lane+32] = (f16)((a1-mu)*rs*g[lane+32] + bb[lane+32]);
  out[row*HC + lane+64] = (f16)((a2-mu)*rs*g[lane+64] + bb[lane+64]);
}
__global__ void k_ln96_f32(const float* __restrict__ X, const float* __restrict__ g,
                           const float* __restrict__ bb, float* __restrict__ out){
  int row  = blockIdx.x*8 + (threadIdx.x >> 5);
  int lane = threadIdx.x & 31;
  const float* xr = X + row*HC;
  float a0 = xr[lane], a1 = xr[lane+32], a2 = xr[lane+64];
  float s = a0+a1+a2, q = a0*a0+a1*a1+a2*a2;
  for (int m = 1; m < 32; m <<= 1){ s += __shfl_xor(s,m,32); q += __shfl_xor(q,m,32); }
  float mu = s*(1.f/HC);
  float rs = rsqrtf(q*(1.f/HC) - mu*mu + 1e-5f);
  out[row*HC + lane]    = (a0-mu)*rs*g[lane]    + bb[lane];
  out[row*HC + lane+32] = (a1-mu)*rs*g[lane+32] + bb[lane+32];
  out[row*HC + lane+64] = (a2-mu)*rs*g[lane+64] + bb[lane+64];
}

// ---------------------------------------------------------------------------
// GEMM kernels (one wave per 16x16 tile)
// ---------------------------------------------------------------------------
// xz[8192,384] = h16[8192,96] * in_proj_w[384,96]^T
__global__ void k_gemm_inproj(const f16* __restrict__ A, const f16* __restrict__ Bw,
                              float* __restrict__ Cout){
  int nt = blockIdx.x, mt = blockIdx.y;
  v8f acc = wmma_tile(A + (size_t)mt*16*HC, HC, Bw + (size_t)nt*16*HC, HC, HC);
  int lane = threadIdx.x & 31; int n = lane & 15; int hi = lane >> 4;
#pragma unroll
  for (int v = 0; v < 8; ++v)
    Cout[(size_t)(mt*16 + v + 8*hi)*384 + nt*16 + n] = acc[v];
}

// xdbl[b][k][l][48] = S_{k&1}[b*L+l', 192] * xpw[k][48,192]^T  (l flipped for k>=2)
__global__ void k_gemm_xproj(const f16* __restrict__ S0, const f16* __restrict__ S1,
                             const f16* __restrict__ Wp, float* __restrict__ xdbl){
  int nt = blockIdx.x, mt = blockIdx.y, k = blockIdx.z;
  const f16* A  = ((k & 1) ? S1 : S0) + (size_t)mt*16*DI;
  const f16* Bw = Wp + ((size_t)k*48 + nt*16)*DI;
  v8f acc = wmma_tile(A, DI, Bw, DI, DI);
  int lane = threadIdx.x & 31; int n = lane & 15; int hi = lane >> 4;
#pragma unroll
  for (int v = 0; v < 8; ++v){
    int gr = mt*16 + v + 8*hi;
    int b = gr >> 12; int lp = gr & (LL-1);
    int l = (k < 2) ? lp : (LL-1-lp);
    xdbl[((size_t)(b*KD + k)*LL + l)*48 + nt*16 + n] = acc[v];
  }
}

// xc[8192,96] = xf*skip1 + y16[8192,192] * out_proj_w[96,192]^T
__global__ void k_gemm_outproj(const f16* __restrict__ Y, const f16* __restrict__ W,
                               const float* __restrict__ xf, const float* __restrict__ skip1,
                               float* __restrict__ xc){
  int nt = blockIdx.x, mt = blockIdx.y;
  v8f acc = wmma_tile(Y + (size_t)mt*16*DI, DI, W + (size_t)nt*16*DI, DI, DI);
  int lane = threadIdx.x & 31; int n = lane & 15; int hi = lane >> 4;
#pragma unroll
  for (int v = 0; v < 8; ++v){
    int r = mt*16 + v + 8*hi;
    int c = nt*16 + n;
    xc[(size_t)r*HC + c] = xf[(size_t)r*HC + c]*skip1[c] + acc[v];
  }
}

// ---------------------------------------------------------------------------
// depthwise 3x3 + bias + SiLU; also emits S0 (l-major) and S1 (transposed) f16
// ---------------------------------------------------------------------------
__global__ void k_dwconv(const float* __restrict__ xz, const float* __restrict__ cw,
                         const float* __restrict__ cb, float* __restrict__ u,
                         f16* __restrict__ S0, f16* __restrict__ S1){
  int i = blockIdx.x*blockDim.x + threadIdx.x;
  if (i >= NB*DI*LL) return;
  int l = i & (LL-1); int d = (i >> 12) % DI; int b = i / (DI*LL);
  int h = l >> 6, w = l & 63;
  float s = cb[d];
#pragma unroll
  for (int dh = -1; dh <= 1; ++dh){
    int hh = h + dh; if (hh < 0 || hh >= HH) continue;
#pragma unroll
    for (int dw = -1; dw <= 1; ++dw){
      int ww = w + dw; if (ww < 0 || ww >= WV) continue;
      s += xz[(size_t)(b*LL + hh*WV + ww)*384 + d] * cw[d*9 + (dh+1)*3 + (dw+1)];
    }
  }
  s = silu_f(s);
  u[(size_t)(b*DI + d)*LL + l] = s;
  S0[(size_t)(b*LL + l)*DI + d] = (f16)s;
  int l1 = w*HH + h;
  S1[(size_t)(b*LL + l1)*DI + d] = (f16)s;
}

// ---------------------------------------------------------------------------
// selective scan: lane = (channel pair, n); 16 lanes per (b,k,d); h in regs.
// delta computed in-loop (rank-6 dot + softplus); y reduced via shfl_xor.
// ---------------------------------------------------------------------------
__global__ void k_scan(const float* __restrict__ xdbl, const float* __restrict__ u,
                       const float* __restrict__ dtw, const float* __restrict__ dtb,
                       const float* __restrict__ Alog, const float* __restrict__ Dsk,
                       float* __restrict__ oy){
  int t32 = blockIdx.x*blockDim.x + threadIdx.x;
  int gch = t32 >> 4;
  int n   = t32 & 15;
  int b = gch / (KD*DI);
  int r = gch % (KD*DI);
  int k = r / DI;
  int d = r % DI;
  float Aeff = -expf(Alog[(k*DI + d)*NS + n]);
  float dsv  = Dsk[k*DI + d];
  const float* wv = dtw + (k*DI + d)*RK;
  float w0=wv[0], w1=wv[1], w2=wv[2], w3=wv[3], w4=wv[4], w5=wv[5];
  float bdt = dtb[k*DI + d];
  const float* xrow = xdbl + (size_t)(b*KD + k)*LL*48;
  const float* ub   = u + (size_t)(b*DI + d)*LL;
  float* orow = oy + ((size_t)(b*KD + k)*DI + d)*LL;
  float hst = 0.f;
  for (int t = 0; t < LL; ++t){
    const float* xr = xrow + t*48;
    float dt = bdt + xr[0]*w0 + xr[1]*w1 + xr[2]*w2 + xr[3]*w3 + xr[4]*w4 + xr[5]*w5;
    float delta = softplus_f(dt);
    float Bv = xr[RK + n];
    float Cv = xr[RK + NS + n];
    int tl;
    if      (k == 0) tl = t;
    else if (k == 1) tl = (t & 63)*WV + (t >> 6);
    else if (k == 2) tl = LL - 1 - t;
    else { int tp = LL - 1 - t; tl = (tp & 63)*WV + (tp >> 6); }
    float ut = ub[tl];
    float dA = expf(delta * Aeff);
    hst = dA*hst + (delta*ut)*Bv;
    float yp = hst * Cv;
    yp += __shfl_xor(yp, 1, 32);
    yp += __shfl_xor(yp, 2, 32);
    yp += __shfl_xor(yp, 4, 32);
    yp += __shfl_xor(yp, 8, 32);
    if (n == 0) orow[t] = yp + dsv*ut;
  }
}

// ---------------------------------------------------------------------------
// combine 4 directions + out_norm (over Di=192) + SiLU(z) gate -> f16
// ---------------------------------------------------------------------------
__global__ void k_combine(const float* __restrict__ oy, const float* __restrict__ xz,
                          const float* __restrict__ g, const float* __restrict__ bb,
                          f16* __restrict__ y16){
  __shared__ float ssum[6], ssq[6], stat[2];
  int bl = blockIdx.x; int b = bl >> 12; int l = bl & (LL-1);
  int d = threadIdx.x;
  int h = l >> 6, w = l & 63;
  int t1 = w*HH + h;
  const float* base = oy + (size_t)b*KD*DI*LL;
  float yv = base[(size_t)(0*DI + d)*LL + l]
           + base[(size_t)(2*DI + d)*LL + (LL-1-l)]
           + base[(size_t)(1*DI + d)*LL + t1]
           + base[(size_t)(3*DI + d)*LL + (LL-1-t1)];
  float s = yv, q = yv*yv;
  for (int m = 1; m < 32; m <<= 1){ s += __shfl_xor(s,m,32); q += __shfl_xor(q,m,32); }
  int wid = threadIdx.x >> 5;
  if ((threadIdx.x & 31) == 0){ ssum[wid] = s; ssq[wid] = q; }
  __syncthreads();
  if (threadIdx.x == 0){
    float S = 0.f, Q = 0.f;
    for (int j = 0; j < 6; ++j){ S += ssum[j]; Q += ssq[j]; }
    float mu = S*(1.f/DI);
    stat[0] = mu;
    stat[1] = rsqrtf(Q*(1.f/DI) - mu*mu + 1e-5f);
  }
  __syncthreads();
  float ln = (yv - stat[0])*stat[1]*g[d] + bb[d];
  float z  = xz[(size_t)(b*LL + l)*384 + DI + d];
  y16[(size_t)(b*LL + l)*DI + d] = (f16)(ln * silu_f(z));
}

// ---------------------------------------------------------------------------
// CAB: conv3x3 96->32 + GELU; conv3x3 32->96; global pool; SE attention; final
// ---------------------------------------------------------------------------
__global__ void k_cab1(const float* __restrict__ y2, const float* __restrict__ w1,
                       const float* __restrict__ b1, float* __restrict__ t1o){
  int i = blockIdx.x*blockDim.x + threadIdx.x;
  if (i >= NB*32*LL) return;
  int l = i & (LL-1); int o = (i >> 12) % 32; int b = i / (32*LL);
  int h = l >> 6, w = l & 63;
  float s = b1[o];
  for (int dh = -1; dh <= 1; ++dh){
    int hh = h + dh; if (hh < 0 || hh >= HH) continue;
    for (int dw = -1; dw <= 1; ++dw){
      int ww = w + dw; if (ww < 0 || ww >= WV) continue;
      const float* xin = y2 + (size_t)(b*LL + hh*WV + ww)*HC;
      const float* wr  = w1 + (size_t)o*HC*9 + (dh+1)*3 + (dw+1);
#pragma unroll 4
      for (int ci = 0; ci < HC; ++ci) s += xin[ci]*wr[ci*9];
    }
  }
  t1o[(size_t)(b*LL + l)*32 + o] = gelu_f(s);
}

__global__ void k_cab2(const float* __restrict__ t1i, const float* __restrict__ w2,
                       const float* __restrict__ b2, float* __restrict__ t2o){
  int i = blockIdx.x*blockDim.x + threadIdx.x;
  if (i >= NB*HC*LL) return;
  int l = i & (LL-1); int o = (i >> 12) % HC; int b = i / (HC*LL);
  int h = l >> 6, w = l & 63;
  float s = b2[o];
  for (int dh = -1; dh <= 1; ++dh){
    int hh = h + dh; if (hh < 0 || hh >= HH) continue;
    for (int dw = -1; dw <= 1; ++dw){
      int ww = w + dw; if (ww < 0 || ww >= WV) continue;
      const float* xin = t1i + (size_t)(b*LL + hh*WV + ww)*32;
      const float* wr  = w2 + (size_t)o*32*9 + (dh+1)*3 + (dw+1);
#pragma unroll 4
      for (int ci = 0; ci < 32; ++ci) s += xin[ci]*wr[ci*9];
    }
  }
  t2o[(size_t)(b*LL + l)*HC + o] = s;
}

__global__ void k_pool(const float* __restrict__ t2, float* __restrict__ p){
  __shared__ float sh[8];
  int c = blockIdx.x % HC; int b = blockIdx.x / HC;
  float s = 0.f;
  for (int l = threadIdx.x; l < LL; l += 256)
    s += t2[(size_t)(b*LL + l)*HC + c];
  for (int m = 1; m < 32; m <<= 1) s += __shfl_xor(s, m, 32);
  int wid = threadIdx.x >> 5;
  if ((threadIdx.x & 31) == 0) sh[wid] = s;
  __syncthreads();
  if (threadIdx.x == 0){
    float S = 0.f; for (int j = 0; j < 8; ++j) S += sh[j];
    p[blockIdx.x] = S * (1.f/LL);
  }
}

__global__ void k_attn(const float* __restrict__ p, const float* __restrict__ dw,
                       const float* __restrict__ db, const float* __restrict__ uw,
                       const float* __restrict__ ub, float* __restrict__ attn){
  int i = threadIdx.x;
  if (i >= NB*HC) return;
  int c = i % HC; int b = i / HC;
  float s = ub[c];
  for (int j = 0; j < 3; ++j){
    float hsum = db[j];
    for (int ci = 0; ci < HC; ++ci) hsum += dw[j*HC + ci]*p[b*HC + ci];
    hsum = fmaxf(hsum, 0.f);
    s += uw[c*3 + j]*hsum;
  }
  attn[i] = sigm_f(s);
}

__global__ void k_final(const float* __restrict__ xc, const float* __restrict__ t2,
                        const float* __restrict__ attn, const float* __restrict__ skip2,
                        float* __restrict__ xfo){
  int i = blockIdx.x*blockDim.x + threadIdx.x;
  if (i >= BL*HC) return;
  int c = i % HC; int bl = i / HC; int b = bl >> 12;
  xfo[i] = xc[i]*skip2[c] + t2[i]*attn[b*HC + c];
}

// ---------------------------------------------------------------------------
extern "C" void kernel_launch(void* const* d_in, const int* in_sizes, int n_in,
                              void* d_out, int out_size, void* d_ws, size_t ws_size,
                              hipStream_t stream) {
  (void)in_sizes; (void)n_in; (void)out_size; (void)ws_size;
  const float* x        = (const float*)d_in[0];
  const float* ln1_g    = (const float*)d_in[1];
  const float* ln1_b    = (const float*)d_in[2];
  const float* in_w     = (const float*)d_in[3];
  const float* conv_w   = (const float*)d_in[4];
  const float* conv_b   = (const float*)d_in[5];
  const float* xp_w     = (const float*)d_in[6];
  const float* dtp_w    = (const float*)d_in[7];
  const float* dtp_b    = (const float*)d_in[8];
  const float* A_logs   = (const float*)d_in[9];
  const float* Ds       = (const float*)d_in[10];
  const float* on_g     = (const float*)d_in[11];
  const float* on_b     = (const float*)d_in[12];
  const float* op_w     = (const float*)d_in[13];
  const float* skip1    = (const float*)d_in[14];
  const float* ln2_g    = (const float*)d_in[15];
  const float* ln2_b    = (const float*)d_in[16];
  const float* cab1_w   = (const float*)d_in[17];
  const float* cab1_b   = (const float*)d_in[18];
  const float* cab2_w   = (const float*)d_in[19];
  const float* cab2_b   = (const float*)d_in[20];
  const float* ca_dw    = (const float*)d_in[21];
  const float* ca_db    = (const float*)d_in[22];
  const float* ca_uw    = (const float*)d_in[23];
  const float* ca_ub    = (const float*)d_in[24];
  const float* skip2    = (const float*)d_in[25];

  char* ws = (char*)d_ws;
  size_t off = 0;
  auto carve = [&](size_t bytes)->void* {
    void* p = ws + off;
    off += (bytes + 255) & ~(size_t)255;
    return p;
  };
  float* xfA   = (float*)carve((size_t)BL*HC*4);
  float* xfB   = (float*)carve((size_t)BL*HC*4);
  f16*   h16   = (f16*)  carve((size_t)BL*HC*2);
  float* xz    = (float*)carve((size_t)BL*384*4);
  float* u     = (float*)carve((size_t)NB*DI*LL*4);
  f16*   S0    = (f16*)  carve((size_t)BL*DI*2);
  f16*   S1    = (f16*)  carve((size_t)BL*DI*2);
  float* xdbl  = (float*)carve((size_t)NB*KD*LL*48*4);
  float* oy    = (float*)carve((size_t)NB*KD*DI*LL*4);
  f16*   y16   = (f16*)  carve((size_t)BL*DI*2);
  float* xc    = (float*)carve((size_t)BL*HC*4);
  float* y2    = (float*)carve((size_t)BL*HC*4);
  float* t1    = (float*)carve((size_t)BL*32*4);
  float* t2    = (float*)carve((size_t)BL*HC*4);
  float* pool  = (float*)carve((size_t)NB*HC*4);
  float* attn  = (float*)carve((size_t)NB*HC*4);
  f16*   wip16 = (f16*)  carve((size_t)384*HC*2);
  f16*   wxp16 = (f16*)  carve((size_t)KD*48*DI*2);
  f16*   wop16 = (f16*)  carve((size_t)HC*DI*2);

  k_nchw_to_nlc<<<(NB*HC*LL + 255)/256, 256, 0, stream>>>(x, xfA);

  float* cur = xfA;
  float* nxt = xfB;
  for (int i = 0; i < 2; ++i) {
    // weight conversion (f32 -> f16)
    k_cvt_f16<<<(384*HC + 255)/256, 256, 0, stream>>>(in_w + (size_t)i*384*HC, wip16, 384*HC);
    k_cvt_xpw<<<(KD*48*DI + 255)/256, 256, 0, stream>>>(xp_w + (size_t)i*KD*(RK+2*NS)*DI, wxp16);
    k_cvt_f16<<<(HC*DI + 255)/256, 256, 0, stream>>>(op_w + (size_t)i*HC*DI, wop16, HC*DI);

    // LN1 -> f16
    k_ln96_f16<<<BL/8, 256, 0, stream>>>(cur, ln1_g + i*HC, ln1_b + i*HC, h16);
    // in_proj: 8192x96 x (384x96)^T  (WMMA)
    k_gemm_inproj<<<dim3(384/16, BL/16), 32, 0, stream>>>(h16, wip16, xz);
    // depthwise conv + SiLU; emits u, S0, S1
    k_dwconv<<<(NB*DI*LL + 255)/256, 256, 0, stream>>>(xz, conv_w + (size_t)i*DI*9,
                                                       conv_b + i*DI, u, S0, S1);
    // x_proj for 4 directions: 8192x192 x (48x192)^T, flip fused (WMMA)
    k_gemm_xproj<<<dim3(48/16, BL/16, KD), 32, 0, stream>>>(S0, S1, wxp16, xdbl);
    // selective scan (dt_proj + softplus fused)
    k_scan<<<(NB*KD*DI*NS)/256, 256, 0, stream>>>(xdbl, u,
        dtp_w + (size_t)i*KD*DI*RK, dtp_b + (size_t)i*KD*DI,
        A_logs + (size_t)i*KD*DI*NS, Ds + (size_t)i*KD*DI, oy);
    // combine directions + out_norm + gate -> f16
    k_combine<<<BL, DI, 0, stream>>>(oy, xz, on_g + i*DI, on_b + i*DI, y16);
    // out_proj + residual*skip1 (WMMA)
    k_gemm_outproj<<<dim3(HC/16, BL/16), 32, 0, stream>>>(y16, wop16, cur,
                                                          skip1 + i*HC, xc);
    // LN2
    k_ln96_f32<<<BL/8, 256, 0, stream>>>(xc, ln2_g + i*HC, ln2_b + i*HC, y2);
    // CAB
    k_cab1<<<(NB*32*LL + 255)/256, 256, 0, stream>>>(y2, cab1_w + (size_t)i*32*HC*9,
                                                     cab1_b + i*32, t1);
    k_cab2<<<(NB*HC*LL + 255)/256, 256, 0, stream>>>(t1, cab2_w + (size_t)i*HC*32*9,
                                                     cab2_b + i*HC, t2);
    k_pool<<<NB*HC, 256, 0, stream>>>(t2, pool);
    k_attn<<<1, 256, 0, stream>>>(pool, ca_dw + (size_t)i*3*HC, ca_db + i*3,
                                  ca_uw + (size_t)i*HC*3, ca_ub + i*HC, attn);
    k_final<<<(BL*HC + 255)/256, 256, 0, stream>>>(xc, t2, attn, skip2 + i*HC, nxt);

    float* tmp = cur; cur = nxt; nxt = tmp;
  }

  k_nlc_to_nchw<<<(NB*HC*LL + 255)/256, 256, 0, stream>>>(cur, (float*)d_out);
}
